// RepMetLoss_56770877719136
// MI455X (gfx1250) — compile-verified, matrix operand order checked by reading, not run on hardware
//
#include <hip/hip_runtime.h>

// ---------------- problem constants (match reference) ----------------
#define NS      8192      // samples
#define NCLS    1000      // classes
#define KREP    5         // reps per class
#define NREP    5000      // N*K
#define EMB     512
#define NTILES  313       // ceil(5000/16)

// ---------------- workspace layout (bytes) ----------------
#define XX_OFF   0                      // 8192 f32
#define RR_OFF   32768                  // 5000 f32
#define PART_OFF 53248                  // 256*2 f32 partials
#define B_OFF    55296                  // swizzled bf16 hi/lo B fragments: 313*16*32*16 dwords (~9.8MB)

typedef __attribute__((ext_vector_type(16))) __bf16       v16bf;
typedef __attribute__((ext_vector_type(8)))  float        v8f;
typedef __attribute__((ext_vector_type(8)))  unsigned int v8u;

static __device__ __forceinline__ v16bf as_bf16(v8u v) {
  return __builtin_bit_cast(v16bf, v);
}
static __device__ __forceinline__ float bf16_round(float a) {
  return (float)(__bf16)a;
}
static __device__ __forceinline__ unsigned int pack_bf16(float a, float b) {
  unsigned short ha = __builtin_bit_cast(unsigned short, (__bf16)a);
  unsigned short hb = __builtin_bit_cast(unsigned short, (__bf16)b);
  return (unsigned int)ha | ((unsigned int)hb << 16);
}
static __device__ __forceinline__ v8f wmma_bf16(v16bf a, v16bf b, v8f c) {
  // D = A(16x32 bf16) x B(32x16 bf16) + C(16x16 f32)
  return __builtin_amdgcn_wmma_f32_16x16x32_bf16(false, a, false, b, (short)0, c, false, false);
}
// 16-bit A/B fragment K-pair base for dword v (0..7), half = lane>>4
static __device__ __forceinline__ int kpair_base(int v, int half) {
  return (v < 4 ? 2 * v : 8 + 2 * v) + 8 * half;
}

// ---------------- kernel 1: row norms of x and reps ----------------
__global__ __launch_bounds__(256) void repmet_norms(const float* __restrict__ x,
                                                    const float* __restrict__ reps,
                                                    float* __restrict__ xx,
                                                    float* __restrict__ rr) {
  int wid = (int)((blockIdx.x * blockDim.x + threadIdx.x) >> 5);
  int lane = threadIdx.x & 31;
  const float* src;
  float* dst;
  if (wid < NS) {
    src = x + (size_t)wid * EMB;
    dst = xx + wid;
  } else {
    int r = wid - NS;
    if (r >= NREP) return;
    src = reps + (size_t)r * EMB;
    dst = rr + r;
  }
  float s = 0.0f;
  for (int i = lane; i < EMB; i += 32) {
    float v = src[i];
    s = fmaf(v, v, s);
  }
#pragma unroll
  for (int off = 16; off > 0; off >>= 1) s += __shfl_xor(s, off, 32);
  if (lane == 0) *dst = s;
}

// ---------------- kernel 2: pre-swizzle reps into bf16 hi/lo B fragments ----------------
// Fragment f = t*16 + kt covers B tile (K = kt*32..kt*32+31, j = t*16..t*16+15).
// Layout: bfr[f][lane][0..7]=hi dwords, [8..15]=lo dwords (each dword = 2 packed bf16).
__global__ __launch_bounds__(128) void repmet_bswizzle(const float* __restrict__ reps,
                                                       unsigned int* __restrict__ bfr) {
  int f = blockIdx.x * 4 + (threadIdx.x >> 5);
  if (f >= NTILES * 16) return;
  int lane = threadIdx.x & 31;
  int t = f >> 4, kt = f & 15;
  int half = lane >> 4, n = lane & 15;
  int j = t * 16 + n;
  size_t base = (size_t)f * 512 + (size_t)lane * 16;
#pragma unroll
  for (int v = 0; v < 8; ++v) {
    int k = kt * 32 + kpair_base(v, half);
    float a = 0.0f, b = 0.0f;
    if (j < NREP) {
      const float* rp = reps + (size_t)j * EMB + k;
      a = rp[0];
      b = rp[1];
    }
    bfr[base + v]     = pack_bf16(a, b);
    bfr[base + 8 + v] = pack_bf16(a - bf16_round(a), b - bf16_round(b));
  }
}

// ---------------- kernel 3: fused split-bf16 GEMM + streamed loss reductions ----------------
// 128 threads (4 waves). Block owns 32 rows (2 M-tiles). Waves split the j (rep)
// dimension at class- and tile-aligned boundaries. Per-row loss state streamed per class.
__global__ __launch_bounds__(128) void repmet_main(const float* __restrict__ x,
                                                   const int* __restrict__ target,
                                                   const float* __restrict__ xx,
                                                   const float* __restrict__ rr,
                                                   const unsigned int* __restrict__ bfr,
                                                   float* __restrict__ part) {
  __shared__ __attribute__((aligned(32))) unsigned int ldsA[2 * 16 * 32 * 16]; // 64 KB A frags
  __shared__ float ldsD2[4][2][16][16];                                        // per-wave d2 tiles
  __shared__ float redS[4][32], redMA[4][32], redMI[4][32], redMC[4][32], redPS[4][32];

  const int tid = threadIdx.x;
  const int w = tid >> 5;
  const int lane = tid & 31;
  const int half = lane >> 4;
  const int nn = lane & 15;
  const int row0 = blockIdx.x * 32;

  // ---- stage x rows as bf16 hi/lo A fragments in LDS (shared by all 4 waves) ----
  for (int it = tid; it < 8192; it += 128) {
    int mt = it >> 12;          // M tile 0/1
    int rem = it & 4095;
    int kt = rem >> 8;          // k-tile 0..15
    int rem2 = rem & 255;
    int ln = rem2 >> 3;         // lane 0..31
    int v = rem2 & 7;           // dword 0..7
    int k = kt * 32 + kpair_base(v, ln >> 4);
    const float* xp = x + (size_t)(row0 + mt * 16 + (ln & 15)) * EMB + k;
    float a = xp[0], b = xp[1];
    int base = ((mt * 16 + kt) * 32 + ln) * 16;
    ldsA[base + v]     = pack_bf16(a, b);
    ldsA[base + 8 + v] = pack_bf16(a - bf16_round(a), b - bf16_round(b));
  }
  __syncthreads();

  // per-lane ||x||^2 for the C-fragment rows it owns
  float xq0[8], xq1[8];
#pragma unroll
  for (int r = 0; r < 8; ++r) {
    xq0[r] = xx[row0 + r + 8 * half];
    xq1[r] = xx[row0 + 16 + r + 8 * half];
  }
  // lane -> block row = lane (scan mapping: tile = lane>>4, row-in-tile = lane&15)
  const int t_row = target[row0 + lane];

  // streamed per-row loss state (class groups are contiguous in j)
  float cur_min = __builtin_inff(), sum_e = 0.0f, min_all = __builtin_inff();
  float min_inc = __builtin_inff(), min_cor = __builtin_inff(), p_sel = 0.0f;
  int cur_cls = -1;

  auto fold = [&]() {
    if (cur_cls >= 0) {
      float p = __expf(-2.0f * cur_min);     // hard prob of finished class
      sum_e += __expf(p);                    // softmax denominator term (logit = p)
      min_all = fminf(min_all, cur_min);
      if (cur_cls == t_row) min_cor = cur_min;
      else min_inc = fminf(min_inc, cur_min);
      if (cur_cls == t_row - 1) p_sel = p;   // CE numerator logit (probs_back[t], t>=1)
    }
  };

  // wave j-chunks: boundaries at tiles 80/160/240 -> j=1280/2560/3840 (class & tile aligned)
  const int tBounds[5] = {0, 80, 160, 240, NTILES};
  const int tBeg = tBounds[w], tEnd = tBounds[w + 1];

  for (int tt = tBeg; tt < tEnd; ++tt) {
    const int jstart = tt * 16;
    v8f acc0 = {};
    v8f acc1 = {};
    const unsigned int* bp = bfr + (size_t)tt * 8192 + (size_t)lane * 16;
#pragma unroll 4
    for (int kt = 0; kt < 16; ++kt) {
      const unsigned int* ap = &ldsA[kt * 512 + lane * 16];
      v8u a0h = *(const v8u*)(ap);
      v8u a0l = *(const v8u*)(ap + 8);
      v8u a1h = *(const v8u*)(ap + 8192);
      v8u a1l = *(const v8u*)(ap + 8192 + 8);
      v8u bh  = *(const v8u*)(bp);
      v8u bl  = *(const v8u*)(bp + 8);
      // split-bf16: dot ~= hi*hi + hi*lo + lo*hi (f32 accumulate)
      acc0 = wmma_bf16(as_bf16(a0h), as_bf16(bh), acc0);
      acc1 = wmma_bf16(as_bf16(a1h), as_bf16(bh), acc1);
      acc0 = wmma_bf16(as_bf16(a0h), as_bf16(bl), acc0);
      acc1 = wmma_bf16(as_bf16(a1h), as_bf16(bl), acc1);
      acc0 = wmma_bf16(as_bf16(a0l), as_bf16(bh), acc0);
      acc1 = wmma_bf16(as_bf16(a1l), as_bf16(bh), acc1);
      bp += 512;
    }
    // ---- epilogue: d2 = ||x||^2 + ||r||^2 - 2*dot -> LDS tile ----
    {
      int j = jstart + nn;
      float rrv = (j < NREP) ? rr[j] : 0.0f;
#pragma unroll
      for (int r = 0; r < 8; ++r) {
        int m = r + 8 * half;
        ldsD2[w][0][m][nn] = xq0[r] + rrv - 2.0f * acc0[r];
        ldsD2[w][1][m][nn] = xq1[r] + rrv - 2.0f * acc1[r];
      }
    }
    __asm__ volatile("s_wait_dscnt 0" ::: "memory"); // same-wave LDS RAW ordering
    // ---- streamed grouped-min scan over this tile's 16 columns ----
    {
      int rem = NREP - jstart;
      const int lim = rem < 16 ? rem : 16;
      for (int col = 0; col < lim; ++col) {
        float v = ldsD2[w][half][nn][col];
        int jj = jstart + col;
        int cls = jj / KREP;
        if (cls != cur_cls) {
          fold();
          cur_cls = cls;
          cur_min = __builtin_inff();
        }
        cur_min = fminf(cur_min, v);
      }
    }
  }
  fold(); // last class of this wave's chunk (chunk boundaries are class-aligned)

  // ---- combine 4 waves' partial row-state, finish per-row losses ----
  redS[w][lane] = sum_e;
  redMA[w][lane] = min_all;
  redMI[w][lane] = min_inc;
  redMC[w][lane] = min_cor;
  redPS[w][lane] = p_sel;
  __syncthreads();
  if (w == 0) {
    float S = 0.0f, MA = __builtin_inff(), MI = __builtin_inff(), MC = __builtin_inff(), PS = 0.0f;
#pragma unroll
    for (int q = 0; q < 4; ++q) {
      S += redS[q][lane];
      MA = fminf(MA, redMA[q][lane]);
      MI = fminf(MI, redMI[q][lane]);
      MC = fminf(MC, redMC[q][lane]);
      PS += redPS[q][lane];
    }
    float back_p = 1.0f - __expf(-2.0f * MA);
    float lt = (t_row == 0) ? back_p : PS;                // logit at index target
    float ce = __logf(S + __expf(back_p)) - lt;           // logits in [0,1]: no max-sub needed
    float ld = fmaxf(MC - MI + 1.0f, 0.0f);               // relu(min_cor - min_inc + ALPHA)
    if (t_row == 0) ld = 0.0f;                            // background rows zeroed
#pragma unroll
    for (int off = 16; off > 0; off >>= 1) {
      ld += __shfl_down(ld, off, 32);
      ce += __shfl_down(ce, off, 32);
    }
    if (lane == 0) {
      part[2 * blockIdx.x]     = ld;
      part[2 * blockIdx.x + 1] = ce;
    }
  }
}

// ---------------- kernel 4: deterministic final reduction ----------------
__global__ __launch_bounds__(128) void repmet_finalize(const float* __restrict__ part,
                                                       float* __restrict__ out) {
  __shared__ float sl[128], sc[128];
  int t = threadIdx.x;
  float a = 0.0f, b = 0.0f;
  for (int i = t; i < 256; i += 128) {
    a += part[2 * i];
    b += part[2 * i + 1];
  }
  sl[t] = a;
  sc[t] = b;
  __syncthreads();
  for (int s = 64; s > 0; s >>= 1) {
    if (t < s) {
      sl[t] += sl[t + s];
      sc[t] += sc[t + s];
    }
    __syncthreads();
  }
  if (t == 0) {
    // LOSS_W * (LOSS_DIST_W * mean(losses_la) + LOSS_CLASS_W^2 * mean(ce))
    out[0] = 0.5f * (sl[0] / (float)NS) + 0.25f * (sc[0] / (float)NS);
  }
}

// ---------------- host launcher ----------------
extern "C" void kernel_launch(void* const* d_in, const int* in_sizes, int n_in,
                              void* d_out, int out_size, void* d_ws, size_t ws_size,
                              hipStream_t stream) {
  (void)in_sizes; (void)n_in; (void)out_size; (void)ws_size;
  const float* x      = (const float*)d_in[0];
  const float* reps   = (const float*)d_in[1];
  const int*   target = (const int*)d_in[2];
  float* out = (float*)d_out;
  char* ws = (char*)d_ws;
  float*        xx   = (float*)(ws + XX_OFF);
  float*        rr   = (float*)(ws + RR_OFF);
  float*        part = (float*)(ws + PART_OFF);
  unsigned int* bfr  = (unsigned int*)(ws + B_OFF);

  // 1) row norms: (8192 + 5000) rows, one wave each, 8 waves per block
  repmet_norms<<<(NS + NREP + 7) / 8, 256, 0, stream>>>(x, reps, xx, rr);
  // 2) swizzle reps into bf16 hi/lo WMMA B fragments: 313*16 fragments, 4 per block
  repmet_bswizzle<<<(NTILES * 16 + 3) / 4, 128, 0, stream>>>(reps, bfr);
  // 3) fused GEMM + loss: 8192/32 = 256 blocks
  repmet_main<<<NS / 32, 128, 0, stream>>>(x, target, xx, rr, bfr, part);
  // 4) deterministic scalar reduction
  repmet_finalize<<<1, 128, 0, stream>>>(part, out);
}
